// Model_1340029796809
// MI455X (gfx1250) — matrix operand order repaired and checked by reading, not compiled, not run on hardware
//
#include <hip/hip_runtime.h>
#include <hip/hip_bf16.h>
#include <stdint.h>

// ---------------- problem constants ----------------
#define W_COUNT 50000
#define MAX_OOVS 100
#define EMB 128
#define HID 256
#define BSZ 16
#define LIN 400
#define TLEN 64
#define VEXT (W_COUNT + MAX_OOVS)          // 50100
#define OUT_ELEMS ((size_t)BSZ * (TLEN - 1) * VEXT)

typedef __bf16 v16bf __attribute__((ext_vector_type(16)));
typedef __bf16 v8bf  __attribute__((ext_vector_type(8)));
typedef float  v8f   __attribute__((ext_vector_type(8)));

// ---------------- WMMA tile loaders (16x16x32 bf16; contiguous k order, same in A and B) ----------------
// A: 16x32 bf16, row-major source [M x K]. lane<16: row=lane, K k0..k0+15
//                                          lane>=16: row=lane-16, K k0+16..k0+31
__device__ __forceinline__ v16bf load_A16(const __bf16* A, int lda, int m0, int k0, int lane) {
  int row = m0 + (lane & 15);
  int kb  = k0 + ((lane >> 4) << 4);
  const __bf16* p = A + (size_t)row * lda + kb;
  v8bf lo = *(const v8bf*)(p);
  v8bf hi = *(const v8bf*)(p + 8);
  v16bf r;
#pragma unroll
  for (int i = 0; i < 8; ++i) { r[i] = lo[i]; r[8 + i] = hi[i]; }
  return r;
}
// B: 32x16 bf16 where B[k][n] = W[n][k], W row-major [N x K].
// lane<16: col=n0+lane, K k0..k0+15 ; lane>=16: col=n0+lane-16, K k0+16..k0+31
__device__ __forceinline__ v16bf load_Bw(const __bf16* W, int ldw, int n0, int k0, int lane) {
  int col = n0 + (lane & 15);
  int kb  = k0 + ((lane >> 4) << 4);
  return *(const v16bf*)(W + (size_t)col * ldw + kb);
}

__device__ __forceinline__ float sigf(float x) { return 1.0f / (1.0f + __expf(-x)); }

__device__ __forceinline__ float waveSum(float v) {
#pragma unroll
  for (int o = 16; o > 0; o >>= 1) v += __shfl_xor(v, o, 32);
  return v;
}
__device__ __forceinline__ float waveMax(float v) {
#pragma unroll
  for (int o = 16; o > 0; o >>= 1) v = fmaxf(v, __shfl_xor(v, o, 32));
  return v;
}

// ---------------- utility kernels ----------------
__global__ void k_zero_f32(float* p, int n) {
  for (int i = blockIdx.x * blockDim.x + threadIdx.x; i < n; i += gridDim.x * blockDim.x) p[i] = 0.f;
}

__global__ void k_f32_to_bf16(const float* __restrict__ src, __bf16* __restrict__ dst, int n) {
  for (int i = blockIdx.x * blockDim.x + threadIdx.x; i < n; i += gridDim.x * blockDim.x)
    dst[i] = (__bf16)src[i];
}

// Combined decoder weight: [Wih | Whh] -> (1024 x 384) bf16
__global__ void k_build_decW(const float* __restrict__ Wih, const float* __restrict__ Whh,
                             __bf16* __restrict__ dst) {
  int i = blockIdx.x * blockDim.x + threadIdx.x;
  if (i >= 1024 * 384) return;
  int n = i / 384, k = i % 384;
  float v = (k < 128) ? Wih[n * 128 + k] : Whh[n * 256 + (k - 128)];
  dst[i] = (__bf16)v;
}

// xs_bf[(l*16+b)*128 + e] = embed[unk(inputs[b][l])][e]
__global__ void k_gather_xs(const int* __restrict__ inputs, const float* __restrict__ embed,
                            __bf16* __restrict__ xs) {
  int idx = blockIdx.x * blockDim.x + threadIdx.x;
  if (idx >= LIN * BSZ * EMB) return;
  int r = idx >> 7, e = idx & 127;
  int l = r >> 4, b = r & 15;
  int id = inputs[b * LIN + l];
  if (id >= W_COUNT) id = 0;
  xs[idx] = (__bf16)embed[(size_t)id * EMB + e];
}

// u = Wh_w^T v (512), u2 = Ws_w^T v (256), csc = {v.Wh_b + v_b, v.Ws_b}
__global__ void k_compute_uv(const float* __restrict__ Wh_w, const float* __restrict__ Wh_b,
                             const float* __restrict__ Ws_w, const float* __restrict__ Ws_b,
                             const float* __restrict__ v_w, const float* __restrict__ v_b,
                             float* __restrict__ u, float* __restrict__ u2, float* __restrict__ csc) {
  int j = threadIdx.x;
  if (j < 512) { float s = 0.f; for (int h = 0; h < 256; ++h) s += v_w[h] * Wh_w[h * 512 + j]; u[j] = s; }
  if (j < 256) { float s = 0.f; for (int h = 0; h < 256; ++h) s += v_w[h] * Ws_w[h * 256 + j]; u2[j] = s; }
  if (j == 0) {
    float s = v_b[0]; for (int h = 0; h < 256; ++h) s += v_w[h] * Wh_b[h]; csc[0] = s;
    float s2 = 0.f;   for (int h = 0; h < 256; ++h) s2 += v_w[h] * Ws_b[h]; csc[1] = s2;
  }
}

// ---------------- generic WMMA GEMM:  C[MxN] = A[MxK](bf16) @ W[NxK]^T(bf16) + bias ----------------
__global__ void k_gemm_bf16(const __bf16* __restrict__ A, int lda,
                            const __bf16* __restrict__ W, int ldw,
                            const float* __restrict__ bias,
                            float* __restrict__ Cf, __bf16* __restrict__ Cb, int ldc,
                            int tiles_m, int tiles_n, int K) {
  int wave = threadIdx.x >> 5, lane = threadIdx.x & 31;
  int tile = blockIdx.x * (blockDim.x >> 5) + wave;
  if (tile >= tiles_m * tiles_n) return;
  int m0 = (tile / tiles_n) * 16, n0 = (tile % tiles_n) * 16;
  v8f acc = {};
  for (int k0 = 0; k0 < K; k0 += 32) {
    v16bf a = load_A16(A, lda, m0, k0, lane);
    v16bf b = load_Bw(W, ldw, n0, k0, lane);
    acc = __builtin_amdgcn_wmma_f32_16x16x32_bf16(false, a, false, b, (short)0, acc, false, false);
  }
  int n  = n0 + (lane & 15);
  int mb = m0 + ((lane >> 4) << 3);
  float bv = bias ? bias[n] : 0.f;
  if (Cf) {
#pragma unroll
    for (int v = 0; v < 8; ++v) Cf[(size_t)(mb + v) * ldc + n] = acc[v] + bv;
  } else {
#pragma unroll
    for (int v = 0; v < 8; ++v) Cb[(size_t)(mb + v) * ldc + n] = (__bf16)(acc[v] + bv);
  }
}

// ---------------- encoder recurrence: blockIdx.x = direction (0 fwd, 1 bwd) ----------------
// gates[b][n] = xW[l][b][n] (bias pre-folded) + (h @ Whh^T)[b][n] ; WMMA over LDS bf16 h.
__global__ void k_enc_scan(const float* __restrict__ xWf, const float* __restrict__ xWb,
                           const __bf16* __restrict__ Whhf, const __bf16* __restrict__ Whhb,
                           float* __restrict__ hsf, float* __restrict__ hsb) {
  extern __shared__ char smem[];
  float*  gates = (float*)smem;                 // 16*1024 f32 = 64KB
  __bf16* hbf   = (__bf16*)(smem + 65536);      // 16*256 bf16 = 8KB
  const float*  xW  = blockIdx.x ? xWb : xWf;
  const __bf16* Whh = blockIdx.x ? Whhb : Whhf;
  float* hs = blockIdx.x ? hsb : hsf;
  int tid = threadIdx.x, wave = tid >> 5, lane = tid & 31;

  float creg[4];
#pragma unroll
  for (int i = 0; i < 4; ++i) creg[i] = 0.f;
  for (int idx = tid; idx < 4096; idx += 1024) hbf[idx] = (__bf16)0.f;
  __syncthreads();

  for (int step = 0; step < LIN; ++step) {
    int l = blockIdx.x ? (LIN - 1 - step) : step;
    const float* xWl = xW + (size_t)l * BSZ * 1024;
    // gates via WMMA: 64 N-tiles over 32 waves
    for (int tn = wave; tn < 64; tn += 32) {
      int n0 = tn * 16;
      v8f acc = {};
      for (int k0 = 0; k0 < 256; k0 += 32) {
        v16bf a = load_A16(hbf, 256, 0, k0, lane);
        v16bf b = load_Bw(Whh, 256, n0, k0, lane);
        acc = __builtin_amdgcn_wmma_f32_16x16x32_bf16(false, a, false, b, (short)0, acc, false, false);
      }
      int n  = n0 + (lane & 15);
      int mb = (lane >> 4) << 3;
#pragma unroll
      for (int v = 0; v < 8; ++v)
        gates[(mb + v) * 1024 + n] = acc[v] + xWl[(size_t)(mb + v) * 1024 + n];
    }
    __syncthreads();
    // LSTM pointwise; c lives in registers (stable thread->element mapping)
#pragma unroll
    for (int i = 0; i < 4; ++i) {
      int idx = tid + i * 1024;
      int b = idx >> 8, n = idx & 255;
      float ig = gates[b * 1024 + n],       fg = gates[b * 1024 + 256 + n];
      float gg = gates[b * 1024 + 512 + n], og = gates[b * 1024 + 768 + n];
      float c = sigf(fg) * creg[i] + sigf(ig) * tanhf(gg);
      float h = sigf(og) * tanhf(c);
      creg[i] = c;
      hbf[idx] = (__bf16)h;
      hs[(size_t)l * 4096 + idx] = h;
    }
    __syncthreads();
  }
}

// whv[b*400+l] = enc(b,l)·u + c0   (attention score precompute)
__global__ void k_whv(const float* __restrict__ hsf, const float* __restrict__ hsb,
                      const float* __restrict__ u, const float* __restrict__ csc,
                      float* __restrict__ whv) {
  int r = blockIdx.x * blockDim.x + threadIdx.x;
  if (r >= BSZ * LIN) return;
  int b = r / LIN, l = r % LIN;
  const float* pf = hsf + (size_t)l * 4096 + b * 256;
  const float* pb = hsb + (size_t)l * 4096 + b * 256;
  float s = csc[0];
  for (int j = 0; j < 256; ++j) s += pf[j] * u[j];
  for (int j = 0; j < 256; ++j) s += pb[j] * u[256 + j];
  whv[r] = s;
}

// ---------------- decoder step (single workgroup, 1024 threads) ----------------
__global__ void k_dec_step(int t,
                           const int* __restrict__ target, const float* __restrict__ embed,
                           const float* __restrict__ whv, const float* __restrict__ u2,
                           const float* __restrict__ csc,
                           const __bf16* __restrict__ decW, const float* __restrict__ dec_b,
                           float* __restrict__ hstate, float* __restrict__ cstate,
                           float* __restrict__ coverage, float* __restrict__ attn_g,
                           float* __restrict__ pcopy, float* __restrict__ pgen,
                           const float* __restrict__ hsf, const float* __restrict__ hsb,
                           __bf16* __restrict__ cat_bf,
                           const float* __restrict__ wh_w, const float* __restrict__ wh_b,
                           const float* __restrict__ ws_w, const float* __restrict__ ws_b,
                           const float* __restrict__ wx_w, const float* __restrict__ wx_b,
                           float* __restrict__ covloss) {
  extern __shared__ char smem[];
  float*  buf  = (float*)smem;                       // 16384 f32: gates, then attn[0:6400] + ctx[8192:16384]
  __bf16* xcat = (__bf16*)(smem + 65536);            // 16 x 384 bf16
  float*  hloc = (float*)(smem + 65536 + 12288);     // 16 x 256 f32
  float*  swsv = (float*)(smem + 65536 + 12288 + 16384); // 16 f32
  int tid = threadIdx.x, wave = tid >> 5, lane = tid & 31;

  // P0: build [emb_t | h] bf16
  for (int idx = tid; idx < BSZ * EMB; idx += 1024) {
    int b = idx >> 7, e = idx & 127;
    int id = target[b * TLEN + t];
    xcat[b * 384 + e] = (__bf16)embed[(size_t)id * EMB + e];
  }
  for (int idx = tid; idx < 4096; idx += 1024) {
    int b = idx >> 8, n = idx & 255;
    xcat[b * 384 + 128 + n] = (__bf16)hstate[idx];
  }
  __syncthreads();

  // P1: gates = [x|h] @ decW^T + dec_b via WMMA (K=384)
  for (int tn = wave; tn < 64; tn += 32) {
    int n0 = tn * 16;
    v8f acc = {};
    for (int k0 = 0; k0 < 384; k0 += 32) {
      v16bf a = load_A16(xcat, 384, 0, k0, lane);
      v16bf b = load_Bw(decW, 384, n0, k0, lane);
      acc = __builtin_amdgcn_wmma_f32_16x16x32_bf16(false, a, false, b, (short)0, acc, false, false);
    }
    int n  = n0 + (lane & 15);
    int mb = (lane >> 4) << 3;
#pragma unroll
    for (int v = 0; v < 8; ++v) buf[(mb + v) * 1024 + n] = acc[v] + dec_b[n];
  }
  __syncthreads();

  // P2: LSTM pointwise
  for (int idx = tid; idx < 4096; idx += 1024) {
    int b = idx >> 8, n = idx & 255;
    float ig = buf[b * 1024 + n],       fg = buf[b * 1024 + 256 + n];
    float gg = buf[b * 1024 + 512 + n], og = buf[b * 1024 + 768 + n];
    float c = sigf(fg) * cstate[idx] + sigf(ig) * tanhf(gg);
    float h = sigf(og) * tanhf(c);
    cstate[idx] = c; hstate[idx] = h; hloc[idx] = h;
    cat_bf[b * 768 + n] = (__bf16)h;   // cat = [state | context]
  }
  __syncthreads();

  // P3: ws_v[b] = h·u2 + c1
  if (wave < 16) {
    int b = wave;
    float s = 0.f;
    for (int j = lane; j < 256; j += 32) s += hloc[b * 256 + j] * u2[j];
    s = waveSum(s);
    if (lane == 0) swsv[b] = s + csc[1];
  }
  __syncthreads();

  // P4: attention softmax + coverage (wave b handles row b)
  if (wave < 16) {
    int b = wave;
    float mx = -1e30f;
    for (int l = lane; l < LIN; l += 32) {
      float s = whv[b * LIN + l] + swsv[(b * LIN + l) & 15]; // faithful tile quirk
      buf[b * LIN + l] = s;
      mx = fmaxf(mx, s);
    }
    mx = waveMax(mx);
    float se = 0.f;
    for (int l = lane; l < LIN; l += 32) {
      float e = __expf(buf[b * LIN + l] - mx);
      buf[b * LIN + l] = e; se += e;
    }
    se = waveSum(se);
    float inv = 1.f / se, cls = 0.f;
    for (int l = lane; l < LIN; l += 32) {
      float a = buf[b * LIN + l] * inv;
      buf[b * LIN + l] = a;
      attn_g[b * LIN + l] = a;
      float cov = coverage[b * LIN + l];
      cls += fminf(a, cov);
      coverage[b * LIN + l] = cov + a;
    }
    cls = waveSum(cls);
    if (lane == 0) atomicAdd(covloss, cls);
  }
  __syncthreads();

  // P5: context[b][j] = sum_l attn * encoded ; ctx stored at buf[8192..]
  for (int idx = tid; idx < BSZ * 512; idx += 1024) {
    int b = idx >> 9, j = idx & 511;
    const float* hp = (j < 256) ? (hsf + b * 256 + j) : (hsb + b * 256 + (j - 256));
    float s = 0.f;
    for (int l = 0; l < LIN; ++l) s += buf[b * LIN + l] * hp[(size_t)l * 4096];
    buf[8192 + idx] = s;
    cat_bf[b * 768 + 256 + j] = (__bf16)s;
  }
  __syncthreads();

  // P6: p_gen
  if (wave < 16) {
    int b = wave;
    float s = 0.f;
    for (int j = lane; j < 512; j += 32) s += buf[8192 + b * 512 + j] * wh_w[j];
    for (int j = lane; j < 256; j += 32) s += hloc[b * 256 + j] * ws_w[j];
    int id = target[b * TLEN + t];
    for (int j = lane; j < 128; j += 32) s += embed[(size_t)id * EMB + j] * wx_w[j];
    s = waveSum(s);
    if (lane == 0) {
      float pg = sigf(s + wh_b[0] + ws_b[0] + wx_b[0]);
      pgen[b] = pg; pcopy[b] = 1.f - pg;
    }
  }
}

// ---------------- per-row softmax over vocab + pev write ----------------
__global__ void k_softmax_pev(const float* __restrict__ logits, const float* __restrict__ pgen,
                              float* __restrict__ out, int t) {
  __shared__ float red[32];
  __shared__ float sbc[2];
  int b = blockIdx.x, tid = threadIdx.x, lane = tid & 31, wave = tid >> 5;
  const float* row = logits + (size_t)b * W_COUNT;

  float mx = -1e30f;
  for (int w = tid; w < W_COUNT; w += 1024) mx = fmaxf(mx, row[w]);
  mx = waveMax(mx);
  if (lane == 0) red[wave] = mx;
  __syncthreads();
  if (wave == 0) {
    float v = (lane < 32) ? red[lane] : -1e30f;
    v = waveMax(v);
    if (lane == 0) sbc[0] = v;
  }
  __syncthreads();
  float rmax = sbc[0];

  float se = 0.f;
  for (int w = tid; w < W_COUNT; w += 1024) se += __expf(row[w] - rmax);
  se = waveSum(se);
  if (lane == 0) red[wave] = se;
  __syncthreads();
  if (wave == 0) {
    float v = (lane < 32) ? red[lane] : 0.f;
    v = waveSum(v);
    if (lane == 0) sbc[1] = v;
  }
  __syncthreads();

  float scale = pgen[b] / sbc[1];
  const float base = 1.0f / (float)W_COUNT / 100.0f;
  float* orow = out + ((size_t)b * (TLEN - 1) + t) * VEXT;
  for (int w = tid; w < VEXT; w += 1024)
    orow[w] = base + ((w < W_COUNT) ? __expf(row[w] - rmax) * scale : 0.f);
}

// ---------------- copy-distribution scatter (faithful flat l*B+b indexing) ----------------
__global__ void k_scatter(const int* __restrict__ inputs, const float* __restrict__ attn,
                          const float* __restrict__ pcopy, float* __restrict__ out, int t) {
  int j = blockIdx.x * blockDim.x + threadIdx.x;
  if (j >= LIN * BSZ) return;
  int b = j & 15, l = j >> 4;
  int id = inputs[j];                       // flat[l*B+b] of the (B,L) array
  float val = pcopy[b] * attn[b * LIN + l];
  atomicAdd(out + ((size_t)b * (TLEN - 1) + t) * VEXT + id, val);
}

// ================= host orchestration =================
extern "C" void kernel_launch(void* const* d_in, const int* in_sizes, int n_in,
                              void* d_out, int out_size, void* d_ws, size_t ws_size,
                              hipStream_t stream) {
  (void)in_sizes; (void)n_in; (void)out_size; (void)ws_size;
  const int*   inputs   = (const int*)d_in[0];
  const int*   target   = (const int*)d_in[1];
  const float* embed    = (const float*)d_in[2];
  const float* encfWih  = (const float*)d_in[3];
  const float* encfWhh  = (const float*)d_in[4];
  const float* encf_b   = (const float*)d_in[5];
  const float* encbWih  = (const float*)d_in[6];
  const float* encbWhh  = (const float*)d_in[7];
  const float* encb_b   = (const float*)d_in[8];
  const float* decWih   = (const float*)d_in[9];
  const float* decWhh   = (const float*)d_in[10];
  const float* dec_b    = (const float*)d_in[11];
  const float* Wh_w     = (const float*)d_in[12];
  const float* Wh_b     = (const float*)d_in[13];
  const float* Ws_w     = (const float*)d_in[14];
  const float* Ws_b     = (const float*)d_in[15];
  const float* v_w      = (const float*)d_in[16];
  const float* v_b      = (const float*)d_in[17];
  const float* wh_w     = (const float*)d_in[18];
  const float* wh_b     = (const float*)d_in[19];
  const float* ws_w     = (const float*)d_in[20];
  const float* ws_b     = (const float*)d_in[21];
  const float* wx_w     = (const float*)d_in[22];
  const float* wx_b     = (const float*)d_in[23];
  const float* V1_w     = (const float*)d_in[24];
  const float* V1_b     = (const float*)d_in[25];
  const float* V2_w     = (const float*)d_in[26];
  const float* V2_b     = (const float*)d_in[27];

  char* w = (char*)d_ws; size_t off = 0;
  auto alloc = [&](size_t bytes) -> void* {
    void* p = w + off; off += (bytes + 255) & ~(size_t)255; return p;
  };
  __bf16* encfWih_bf = (__bf16*)alloc((size_t)1024 * 128 * 2);
  __bf16* encfWhh_bf = (__bf16*)alloc((size_t)1024 * 256 * 2);
  __bf16* encbWih_bf = (__bf16*)alloc((size_t)1024 * 128 * 2);
  __bf16* encbWhh_bf = (__bf16*)alloc((size_t)1024 * 256 * 2);
  __bf16* decW_bf    = (__bf16*)alloc((size_t)1024 * 384 * 2);
  __bf16* V1_bf      = (__bf16*)alloc((size_t)768 * 768 * 2);
  __bf16* V2_bf      = (__bf16*)alloc((size_t)W_COUNT * 768 * 2);
  __bf16* xs_bf      = (__bf16*)alloc((size_t)LIN * BSZ * EMB * 2);
  float*  xWf        = (float*)alloc((size_t)LIN * BSZ * 1024 * 4);
  float*  xWb        = (float*)alloc((size_t)LIN * BSZ * 1024 * 4);
  float*  hsf        = (float*)alloc((size_t)LIN * BSZ * HID * 4);
  float*  hsb        = (float*)alloc((size_t)LIN * BSZ * HID * 4);
  float*  whv        = (float*)alloc((size_t)BSZ * LIN * 4);
  float*  uvec       = (float*)alloc(512 * 4);
  float*  u2vec      = (float*)alloc(256 * 4);
  float*  csc        = (float*)alloc(2 * 4);
  float*  hstate     = (float*)alloc((size_t)BSZ * HID * 4);
  float*  cstate     = (float*)alloc((size_t)BSZ * HID * 4);
  float*  coverage   = (float*)alloc((size_t)BSZ * LIN * 4);
  float*  attn_g     = (float*)alloc((size_t)BSZ * LIN * 4);
  float*  pcopy      = (float*)alloc(BSZ * 4);
  float*  pgen       = (float*)alloc(BSZ * 4);
  __bf16* cat_bf     = (__bf16*)alloc((size_t)BSZ * 768 * 2);
  __bf16* hid_bf     = (__bf16*)alloc((size_t)BSZ * 768 * 2);
  float*  logits     = (float*)alloc((size_t)BSZ * W_COUNT * 4);

  float* out = (float*)d_out;
  float* covloss = out + OUT_ELEMS;

  // --- weight conversion (every call; deterministic) ---
  k_f32_to_bf16<<<512, 256, 0, stream>>>(encfWih, encfWih_bf, 1024 * 128);
  k_f32_to_bf16<<<1024, 256, 0, stream>>>(encfWhh, encfWhh_bf, 1024 * 256);
  k_f32_to_bf16<<<512, 256, 0, stream>>>(encbWih, encbWih_bf, 1024 * 128);
  k_f32_to_bf16<<<1024, 256, 0, stream>>>(encbWhh, encbWhh_bf, 1024 * 256);
  k_f32_to_bf16<<<2048, 256, 0, stream>>>(V1_w, V1_bf, 768 * 768);
  k_f32_to_bf16<<<8192, 256, 0, stream>>>(V2_w, V2_bf, W_COUNT * 768);
  k_build_decW<<<(1024 * 384 + 255) / 256, 256, 0, stream>>>(decWih, decWhh, decW_bf);
  k_gather_xs<<<(LIN * BSZ * EMB + 255) / 256, 256, 0, stream>>>(inputs, embed, xs_bf);
  k_compute_uv<<<1, 512, 0, stream>>>(Wh_w, Wh_b, Ws_w, Ws_b, v_w, v_b, uvec, u2vec, csc);

  // --- state init ---
  k_zero_f32<<<16, 256, 0, stream>>>(hstate, BSZ * HID);
  k_zero_f32<<<16, 256, 0, stream>>>(cstate, BSZ * HID);
  k_zero_f32<<<25, 256, 0, stream>>>(coverage, BSZ * LIN);
  k_zero_f32<<<1, 32, 0, stream>>>(covloss, 1);

  // --- encoder input GEMMs: (6400 x 1024, K=128), bias folded ---
  {
    int tiles_m = (LIN * BSZ) / 16, tiles_n = 1024 / 16;   // 400 x 64
    int blocks = (tiles_m * tiles_n + 7) / 8;              // 8 waves/block
    k_gemm_bf16<<<blocks, 256, 0, stream>>>(xs_bf, 128, encfWih_bf, 128, encf_b,
                                            xWf, nullptr, 1024, tiles_m, tiles_n, 128);
    k_gemm_bf16<<<blocks, 256, 0, stream>>>(xs_bf, 128, encbWih_bf, 128, encb_b,
                                            xWb, nullptr, 1024, tiles_m, tiles_n, 128);
  }

  // --- encoder recurrence (one block per direction) ---
  k_enc_scan<<<2, 1024, 65536 + 8192, stream>>>(xWf, xWb, encfWhh_bf, encbWhh_bf, hsf, hsb);

  // --- attention score precompute ---
  k_whv<<<(BSZ * LIN + 255) / 256, 256, 0, stream>>>(hsf, hsb, uvec, csc, whv);

  // --- decoder loop ---
  const size_t dec_smem = 65536 + 12288 + 16384 + 256;
  for (int t = 0; t < TLEN - 1; ++t) {
    k_dec_step<<<1, 1024, dec_smem, stream>>>(t, target, embed, whv, u2vec, csc,
                                              decW_bf, dec_b, hstate, cstate, coverage,
                                              attn_g, pcopy, pgen, hsf, hsb, cat_bf,
                                              wh_w, wh_b, ws_w, ws_b, wx_w, wx_b, covloss);
    // hidden = cat @ V1^T + V1_b  -> bf16 (16 x 768, K=768)
    k_gemm_bf16<<<6, 256, 0, stream>>>(cat_bf, 768, V1_bf, 768, V1_b,
                                       nullptr, hid_bf, 768, 1, 48, 768);
    // logits = hidden @ V2^T + V2_b  (16 x 50000, K=768)
    k_gemm_bf16<<<(3125 + 7) / 8, 256, 0, stream>>>(hid_bf, 768, V2_bf, 768, V2_b,
                                                    logits, nullptr, W_COUNT, 1, 3125, 768);
    k_softmax_pev<<<BSZ, 1024, 0, stream>>>(logits, pgen, out, t);
    k_scatter<<<(LIN * BSZ + 255) / 256, 256, 0, stream>>>(inputs, attn_g, pcopy, out, t);
  }
}
